// TransformerXL_59983513256534
// MI455X (gfx1250) — matrix operand order repaired and checked by reading, not compiled, not run on hardware
//
#include <hip/hip_runtime.h>
#include <hip/hip_bf16.h>
#include <cstdint>

#define QLEN   1024
#define MLEN_  1024
#define KLEN   2048
#define BSZ    2
#define NHEAD  16
#define DHEAD  64
#define DMODEL 1024
#define DINNER 4096

typedef __bf16 bf16_t;
typedef __attribute__((ext_vector_type(16))) __bf16 v16bf;
typedef __attribute__((ext_vector_type(8)))  float  v8f;
typedef __attribute__((ext_vector_type(4)))  int    v4i_t;

#if defined(__has_builtin)
#if __has_builtin(__builtin_amdgcn_global_load_async_to_lds_b128)
#define HAVE_ASYNC_LDS 1
#endif
#endif

// 16B global -> LDS copy. CDNA5 path: GLOBAL_LOAD_ASYNC_TO_LDS_B128 (ASYNCcnt,
// no VGPR round trip); fallback: plain load+store (compiler-managed waits).
// Builtin signature (from clang diagnostic): (v4i AS1*, v4i AS3*, Imm, Imm).
__device__ __forceinline__ void async_copy_b128(const bf16_t* g, bf16_t* l) {
#ifdef HAVE_ASYNC_LDS
  __builtin_amdgcn_global_load_async_to_lds_b128(
      (__attribute__((address_space(1))) v4i_t*)(void*)g,
      (__attribute__((address_space(3))) v4i_t*)(void*)l, 0, 0);
#else
  *(uint4*)l = *(const uint4*)g;
#endif
}
__device__ __forceinline__ void wait_async_le4() {
#ifdef HAVE_ASYNC_LDS
  asm volatile("s_wait_asynccnt 0x4" ::: "memory");
#endif
}
__device__ __forceinline__ void wait_async_le0() {
#ifdef HAVE_ASYNC_LDS
  asm volatile("s_wait_asynccnt 0x0" ::: "memory");
#endif
}

__device__ __forceinline__ v8f wmma_bf16(v16bf a, v16bf b, v8f c) {
  return __builtin_amdgcn_wmma_f32_16x16x32_bf16(false, a, false, b, (short)0, c,
                                                 false, false);
}

// A-matrix 16x32 (MxK) fragment, row-major source with stride ld.
// Per ISA 7.12.2: lanes 0-15 hold K={0..7,16..23}, lanes 16-31 hold K={8..15,24..31}.
__device__ __forceinline__ v16bf load_afrag(const bf16_t* p, int ld) {
  int lane = threadIdx.x & 31;
  int m = lane & 15, h = lane >> 4;
  const bf16_t* row = p + m * ld;
  v16bf f;
#pragma unroll
  for (int e = 0; e < 16; ++e) {
    int k = (e < 8 ? e : e + 8) + (h << 3);
    f[e] = row[k];
  }
  return f;
}

// B-matrix 32x16 (KxN) fragment; source stored as (N,K) row-major (i.e. B^T).
// Lanes 0-15 hold K={0..7,16..23}, lanes 16-31 hold K={8..15,24..31}.
__device__ __forceinline__ v16bf load_bfrag(const bf16_t* p, int ld) {
  int lane = threadIdx.x & 31;
  int n = lane & 15, h = lane >> 4;
  const bf16_t* row = p + n * ld;
  v16bf f;
#pragma unroll
  for (int e = 0; e < 16; ++e) f[e] = row[(e < 8 ? e : e + 8) + (h << 3)];
  return f;
}

// -------------------- generic bf16 GEMM: C = A * B^T (+bias)(+relu) ----------
// A: (M,K) bf16 lda; B: (N,K) bf16 ldb; C: (M,N) f32 ldc; optional bf16 mirror.
// Batched via blockIdx.z: A += z*aBatch, B += (z>>bShift)*bBatch, C += z*cBatch.
// M,N multiples of 128; K multiple of 32. Double-buffered LDS tiles staged with
// async-to-LDS copies overlapped against WMMA on the previous tile.
__global__ __launch_bounds__(256) void gemm_bf16_kernel(
    const bf16_t* __restrict__ A, const bf16_t* __restrict__ B,
    float* __restrict__ C, bf16_t* __restrict__ Cbf,
    const float* __restrict__ bias,
    int K, int lda, int ldb, int ldc,
    long aBatch, long bBatch, int bShift, long cBatch, int relu) {
  __shared__ __align__(16) bf16_t As[2][128 * 32];
  __shared__ __align__(16) bf16_t Bs[2][128 * 32];
  const int z = blockIdx.z;
  A += (long)z * aBatch;
  B += (long)(z >> bShift) * bBatch;
  C += (long)z * cBatch;
  if (Cbf) Cbf += (long)z * cBatch;
  const long bm = (long)blockIdx.y * 128;
  const long bn = (long)blockIdx.x * 128;
  const int tid = threadIdx.x;
  const int lane = tid & 31, wid = tid >> 5;
  const int wm = (wid >> 2) * 64;   // wave row offset in tile (0/64)
  const int wn = (wid & 3) * 32;    // wave col offset (0,32,64,96)
  const v8f vzero = {0.f, 0.f, 0.f, 0.f, 0.f, 0.f, 0.f, 0.f};
  v8f acc[4][2];
#pragma unroll
  for (int t = 0; t < 4; ++t)
#pragma unroll
    for (int u = 0; u < 2; ++u) acc[t][u] = vzero;
  const int lr = tid >> 2;          // 0..63
  const int lc = (tid & 3) * 8;     // 0,8,16,24 (8 bf16 = 16B)
  auto stage = [&](int buf, int k0) {
    async_copy_b128(&A[(bm + lr) * (long)lda + k0 + lc],      &As[buf][lr * 32 + lc]);
    async_copy_b128(&A[(bm + lr + 64) * (long)lda + k0 + lc], &As[buf][(lr + 64) * 32 + lc]);
    async_copy_b128(&B[(bn + lr) * (long)ldb + k0 + lc],      &Bs[buf][lr * 32 + lc]);
    async_copy_b128(&B[(bn + lr + 64) * (long)ldb + k0 + lc], &Bs[buf][(lr + 64) * 32 + lc]);
  };
  const int nk = K >> 5;
  stage(0, 0);
  for (int t = 0; t < nk; ++t) {
    const int cur = t & 1;
    if (t + 1 < nk) {
      stage(cur ^ 1, (t + 1) << 5);   // buffer freed by barrier at end of t-1
      wait_async_le4();               // previous tile's 4 copies complete
    } else {
      wait_async_le0();
    }
    if (t + 2 < nk) {                 // L2 prefetch two tiles ahead
      __builtin_prefetch(&A[(bm + lr) * (long)lda + ((t + 2) << 5) + lc], 0, 1);
      __builtin_prefetch(&B[(bn + lr) * (long)ldb + ((t + 2) << 5) + lc], 0, 1);
    }
    __syncthreads();                  // tile t visible to all waves
    v16bf af[4], bfv[2];
#pragma unroll
    for (int tt = 0; tt < 4; ++tt) af[tt] = load_afrag(&As[cur][(wm + tt * 16) * 32], 32);
#pragma unroll
    for (int u = 0; u < 2; ++u) bfv[u] = load_bfrag(&Bs[cur][(wn + u * 16) * 32], 32);
#pragma unroll
    for (int tt = 0; tt < 4; ++tt)
#pragma unroll
      for (int u = 0; u < 2; ++u) acc[tt][u] = wmma_bf16(af[tt], bfv[u], acc[tt][u]);
    __syncthreads();                  // everyone done reading tile t
  }
  const int nc = lane & 15, hf = lane >> 4;
#pragma unroll
  for (int t = 0; t < 4; ++t)
#pragma unroll
    for (int u = 0; u < 2; ++u) {
      long r0 = bm + wm + t * 16 + hf * 8;
      long c0 = bn + wn + u * 16 + nc;
#pragma unroll
      for (int g = 0; g < 8; ++g) {
        float v = acc[t][u][g];
        if (bias) v += bias[c0];
        if (relu) v = fmaxf(v, 0.f);
        long idx = (r0 + g) * (long)ldc + c0;
        C[idx] = v;
        if (Cbf) Cbf[idx] = (bf16_t)v;
      }
    }
}

// ---------------- fused flash attention with precomputed relative term -------
// z = n*2 + b. Qw/Kh/Vh: [z][seq][64] bf16. QR: [z][1024][2048] f32
// (QR[i][p] = (q_i + r_r_bias) . r_head_k[p]); BD[i,j] = QR[i][j+QLEN-1-i] for
// j <= i + MLEN (the rel_shift wrap region coincides with the causal mask).
// Output avbf: (qlen*bsz, 1024) bf16 rows i*2+b.
__global__ __launch_bounds__(256) void attn_kernel(
    const bf16_t* __restrict__ Qw, const bf16_t* __restrict__ Kh,
    const bf16_t* __restrict__ Vh, const float* __restrict__ QR,
    bf16_t* __restrict__ avbf) {
  __shared__ __align__(16) bf16_t Kt[64 * 64];       // [j][d]
  __shared__ __align__(16) bf16_t Vt[64 * 64];       // [d][j] (transposed)
  __shared__ __align__(16) bf16_t Pl[8][16 * 64];    // per-wave P staging
  const int z = blockIdx.y;
  const int nh = z >> 1, b = z & 1;
  const int tid = threadIdx.x, lane = tid & 31, wid = tid >> 5;
  const int i0 = blockIdx.x * 128 + wid * 16;        // wave's 16-row strip
  const bf16_t* Qb  = Qw + ((long)z * QLEN + i0) * DHEAD;
  const bf16_t* Kz  = Kh + (long)z * KLEN * DHEAD;
  const bf16_t* Vz  = Vh + (long)z * KLEN * DHEAD;
  const float*  QRz = QR + (long)z * QLEN * KLEN;
  const v16bf aq0 = load_afrag(Qb, DHEAD);
  const v16bf aq1 = load_afrag(Qb + 32, DHEAD);
  const v8f vzero = {0.f, 0.f, 0.f, 0.f, 0.f, 0.f, 0.f, 0.f};
  v8f Oacc[4];
#pragma unroll
  for (int u = 0; u < 4; ++u) Oacc[u] = vzero;
  float mrow[8], lrow[8];
#pragma unroll
  for (int g = 0; g < 8; ++g) { mrow[g] = -3.0e38f; lrow[g] = 0.f; }
  const int nc = lane & 15, hf = lane >> 4;
  const int jlimit = blockIdx.x * 128 + 127 + MLEN_;  // block-uniform causal bound
  for (int j0 = 0; j0 < KLEN && j0 <= jlimit; j0 += 64) {
    __syncthreads();
    {   // cooperative K/V chunk load (64x64); K via async-to-LDS, V transposed
      int jr = tid >> 2;
      int c  = (tid & 3) * 16;
      async_copy_b128(&Kz[(long)(j0 + jr) * 64 + c],     &Kt[jr * 64 + c]);
      async_copy_b128(&Kz[(long)(j0 + jr) * 64 + c + 8], &Kt[jr * 64 + c + 8]);
      const bf16_t* vs = &Vz[(long)(j0 + jr) * 64 + c];
#pragma unroll
      for (int e = 0; e < 16; ++e) Vt[(c + e) * 64 + jr] = vs[e];
    }
    wait_async_le0();
    __syncthreads();
    v8f s[4];
#pragma unroll
    for (int jj = 0; jj < 4; ++jj) {   // AC = Q K^T via WMMA
      v8f sa = vzero;
      sa = wmma_bf16(aq0, load_bfrag(&Kt[jj * 16 * 64], 64), sa);
      sa = wmma_bf16(aq1, load_bfrag(&Kt[jj * 16 * 64 + 32], 64), sa);
      s[jj] = sa;
    }
#pragma unroll
    for (int jj = 0; jj < 4; ++jj)     // + BD, scale, mask
#pragma unroll
      for (int g = 0; g < 8; ++g) {
        int ig = i0 + g + hf * 8;
        int jg = j0 + jj * 16 + nc;
        float v;
        if (jg <= ig + MLEN_) {
          float bd = QRz[(long)ig * KLEN + (jg + QLEN - 1 - ig)];
          v = (s[jj][g] + bd) * 0.125f;   // scale = 1/sqrt(64)
        } else {
          v = -3.0e38f;
        }
        s[jj][g] = v;
      }
#pragma unroll
    for (int g = 0; g < 8; ++g) {      // online softmax per row
      float cm = fmaxf(fmaxf(s[0][g], s[1][g]), fmaxf(s[2][g], s[3][g]));
#pragma unroll
      for (int m = 1; m < 16; m <<= 1) cm = fmaxf(cm, __shfl_xor(cm, m, 32));
      float mn = fmaxf(mrow[g], cm);
      float sc = __expf(mrow[g] - mn);
      mrow[g] = mn;
      float rs = 0.f;
#pragma unroll
      for (int jj = 0; jj < 4; ++jj) {
        float p = __expf(s[jj][g] - mn);
        s[jj][g] = p;
        rs += p;
      }
#pragma unroll
      for (int m = 1; m < 16; m <<= 1) rs += __shfl_xor(rs, m, 32);
      lrow[g] = lrow[g] * sc + rs;
#pragma unroll
      for (int u = 0; u < 4; ++u) Oacc[u][g] *= sc;
    }
    bf16_t* pw = &Pl[wid][0];          // stage P (C-layout -> A-layout via LDS)
#pragma unroll
    for (int jj = 0; jj < 4; ++jj)
#pragma unroll
      for (int g = 0; g < 8; ++g)
        pw[(g + hf * 8) * 64 + jj * 16 + nc] = (bf16_t)s[jj][g];
    v16bf ap0 = load_afrag(pw, 64);
    v16bf ap1 = load_afrag(pw + 32, 64);
#pragma unroll
    for (int u = 0; u < 4; ++u) {      // O += P V via WMMA
      Oacc[u] = wmma_bf16(ap0, load_bfrag(&Vt[u * 16 * 64], 64), Oacc[u]);
      Oacc[u] = wmma_bf16(ap1, load_bfrag(&Vt[u * 16 * 64 + 32], 64), Oacc[u]);
    }
  }
#pragma unroll
  for (int u = 0; u < 4; ++u)
#pragma unroll
    for (int g = 0; g < 8; ++g) {
      int ig = i0 + g + hf * 8;
      float v = Oacc[u][g] / lrow[g];
      avbf[((long)ig * BSZ + b) * DMODEL + nh * DHEAD + u * 16 + nc] = (bf16_t)v;
    }
}

// ---------------- residual add + LayerNorm (row = 1024) ----------------------
__global__ __launch_bounds__(256) void add_ln_kernel(
    const float* __restrict__ X, const float* __restrict__ Y,
    const float* __restrict__ G, const float* __restrict__ Bt,
    float* __restrict__ Out, bf16_t* __restrict__ Outbf) {
  __shared__ float red[16];
  const long row = blockIdx.x;
  const int tid = threadIdx.x, lane = tid & 31, wid = tid >> 5;
  const float* xr = X + row * DMODEL;
  const float* yr = Y + row * DMODEL;
  float v[4], s = 0.f, s2 = 0.f;
#pragma unroll
  for (int e = 0; e < 4; ++e) {
    int c = tid + e * 256;
    float t = xr[c] + yr[c];
    v[e] = t; s += t; s2 += t * t;
  }
#pragma unroll
  for (int m = 1; m < 32; m <<= 1) { s += __shfl_xor(s, m, 32); s2 += __shfl_xor(s2, m, 32); }
  if (lane == 0) { red[wid] = s; red[8 + wid] = s2; }
  __syncthreads();
  float ts = 0.f, t2 = 0.f;
#pragma unroll
  for (int w8 = 0; w8 < 8; ++w8) { ts += red[w8]; t2 += red[8 + w8]; }
  const float mu  = ts * (1.f / DMODEL);
  const float var = t2 * (1.f / DMODEL) - mu * mu;
  const float inv = rsqrtf(var + 1e-5f);
#pragma unroll
  for (int e = 0; e < 4; ++e) {
    int c = tid + e * 256;
    float o = (v[e] - mu) * inv * G[c] + Bt[c];
    Out[row * DMODEL + c] = o;
    if (Outbf) Outbf[row * DMODEL + c] = (bf16_t)o;
  }
}

// ---------------- small prep kernels ----------------------------------------
__global__ __launch_bounds__(256) void cast_bf16_kernel(
    const float* __restrict__ src, bf16_t* __restrict__ dst, long n) {
  long i = (long)blockIdx.x * 256 + threadIdx.x;
  if (i < n) dst[i] = (bf16_t)src[i];
}

// w_heads (4096,3072) -> Qw/Qr [z=n*2+b][i][d] (+biases), Kh/Vh [z][kpos][d]
__global__ __launch_bounds__(256) void prep_qkv_kernel(
    const float* __restrict__ WH, const float* __restrict__ rwb,
    const float* __restrict__ rrb, bf16_t* __restrict__ Qw,
    bf16_t* __restrict__ Qr, bf16_t* __restrict__ Kh, bf16_t* __restrict__ Vh) {
  long idx = (long)blockIdx.x * 256 + threadIdx.x;
  int m = (int)(idx / 3072);
  int c = (int)(idx % 3072);
  int kpos = m >> 1, b = m & 1;
  int sec = c >> 10;
  int cc = c & 1023;
  int n = cc >> 6, d = cc & 63;
  float v = WH[idx];
  int z = n * 2 + b;
  if (sec == 0) {
    if (kpos >= MLEN_) {
      int i = kpos - MLEN_;
      long o = ((long)z * QLEN + i) * DHEAD + d;
      Qw[o] = (bf16_t)(v + rwb[n * DHEAD + d]);
      Qr[o] = (bf16_t)(v + rrb[n * DHEAD + d]);
    }
  } else if (sec == 1) {
    Kh[((long)z * KLEN + kpos) * DHEAD + d] = (bf16_t)v;
  } else {
    Vh[((long)z * KLEN + kpos) * DHEAD + d] = (bf16_t)v;
  }
}

// r_head_k (2048,1024) f32 -> Rb [n][j][d] bf16
__global__ __launch_bounds__(256) void prep_r_kernel(
    const float* __restrict__ RHK, bf16_t* __restrict__ Rb) {
  long idx = (long)blockIdx.x * 256 + threadIdx.x;
  int j = (int)(idx >> 10);
  int cc = (int)(idx & 1023);
  int n = cc >> 6, d = cc & 63;
  Rb[((long)n * KLEN + j) * DHEAD + d] = (bf16_t)RHK[idx];
}

// ---------------------------------------------------------------------------
extern "C" void kernel_launch(void* const* d_in, const int* in_sizes, int n_in,
                              void* d_out, int out_size, void* d_ws, size_t ws_size,
                              hipStream_t stream) {
  (void)in_sizes; (void)n_in; (void)out_size; (void)ws_size;
  const float* w      = (const float*)d_in[0];
  const float* r      = (const float*)d_in[1];
  const float* mems   = (const float*)d_in[2];
  /* d_in[3] attn_mask: causal structure recomputed analytically */
  const float* qkv_w  = (const float*)d_in[4];
  const float* rnet_w = (const float*)d_in[5];
  const float* o_w    = (const float*)d_in[6];
  const float* r_r_b  = (const float*)d_in[7];
  const float* r_w_b  = (const float*)d_in[8];
  const float* ln1_g  = (const float*)d_in[9];
  const float* ln1_b  = (const float*)d_in[10];
  const float* ff_w1  = (const float*)d_in[11];
  const float* ff_b1  = (const float*)d_in[12];
  const float* ff_w2  = (const float*)d_in[13];
  const float* ff_b2  = (const float*)d_in[14];
  const float* ln2_g  = (const float*)d_in[15];
  const float* ln2_b  = (const float*)d_in[16];
  float* out = (float*)d_out;

  char* ws = (char*)d_ws;
  size_t off = 0;
  auto alloc = [&](size_t bytes) -> void* {
    void* p = ws + off;
    off += (bytes + 255) & ~(size_t)255;
    return p;
  };
  bf16_t* catbf  = (bf16_t*)alloc((size_t)4096 * 1024 * 2);          // [klen*bsz][dmodel]
  bf16_t* qkvwb  = (bf16_t*)alloc((size_t)3072 * 1024 * 2);
  bf16_t* rbf    = (bf16_t*)alloc((size_t)2048 * 1024 * 2);
  bf16_t* rnwb   = (bf16_t*)alloc((size_t)1024 * 1024 * 2);
  bf16_t* owb    = (bf16_t*)alloc((size_t)1024 * 1024 * 2);
  bf16_t* f1wb   = (bf16_t*)alloc((size_t)4096 * 1024 * 2);
  bf16_t* f2wb   = (bf16_t*)alloc((size_t)1024 * 4096 * 2);
  float*  wheads = (float*)alloc((size_t)4096 * 3072 * 4);
  float*  rhk    = (float*)alloc((size_t)2048 * 1024 * 4);
  bf16_t* Qwb    = (bf16_t*)alloc((size_t)32 * 1024 * 64 * 2);
  bf16_t* Qrb    = (bf16_t*)alloc((size_t)32 * 1024 * 64 * 2);
  bf16_t* Khb    = (bf16_t*)alloc((size_t)32 * 2048 * 64 * 2);
  bf16_t* Vhb    = (bf16_t*)alloc((size_t)32 * 2048 * 64 * 2);
  bf16_t* Rbb    = (bf16_t*)alloc((size_t)16 * 2048 * 64 * 2);
  float*  QRb    = (float*)alloc((size_t)32 * 1024 * 2048 * 4);      // 256 MiB
  bf16_t* avb    = (bf16_t*)alloc((size_t)2048 * 1024 * 2);
  float*  atno   = (float*)alloc((size_t)2048 * 1024 * 4);
  float*  hbuf   = (float*)alloc((size_t)2048 * 1024 * 4);
  bf16_t* hbf    = (bf16_t*)alloc((size_t)2048 * 1024 * 2);
  float*  ff1o   = (float*)alloc((size_t)2048 * 4096 * 4);
  bf16_t* coreb  = (bf16_t*)alloc((size_t)2048 * 4096 * 2);
  float*  ff2o   = (float*)alloc((size_t)2048 * 1024 * 4);

  const dim3 blk(256);
  // f32 -> bf16 staging (cat = [mems; w] along seq)
  cast_bf16_kernel<<<8192, blk, 0, stream>>>(mems, catbf, 2097152);
  cast_bf16_kernel<<<8192, blk, 0, stream>>>(w, catbf + 2097152, 2097152);
  cast_bf16_kernel<<<12288, blk, 0, stream>>>(qkv_w, qkvwb, (long)3072 * 1024);
  cast_bf16_kernel<<<8192, blk, 0, stream>>>(r, rbf, (long)2048 * 1024);
  cast_bf16_kernel<<<4096, blk, 0, stream>>>(rnet_w, rnwb, (long)1024 * 1024);
  cast_bf16_kernel<<<4096, blk, 0, stream>>>(o_w, owb, (long)1024 * 1024);
  cast_bf16_kernel<<<16384, blk, 0, stream>>>(ff_w1, f1wb, (long)4096 * 1024);
  cast_bf16_kernel<<<16384, blk, 0, stream>>>(ff_w2, f2wb, (long)4096 * 1024);

  // w_heads = cat @ qkv_w^T : (4096,3072) K=1024
  gemm_bf16_kernel<<<dim3(24, 32, 1), blk, 0, stream>>>(
      catbf, qkvwb, wheads, nullptr, nullptr, 1024, 1024, 1024, 3072,
      0L, 0L, 0, 0L, 0);
  // r_head_k = r @ r_net_w^T : (2048,1024) K=1024
  gemm_bf16_kernel<<<dim3(8, 16, 1), blk, 0, stream>>>(
      rbf, rnwb, rhk, nullptr, nullptr, 1024, 1024, 1024, 1024,
      0L, 0L, 0, 0L, 0);
  // head split + biases
  prep_qkv_kernel<<<49152, blk, 0, stream>>>(wheads, r_w_b, r_r_b, Qwb, Qrb, Khb, Vhb);
  prep_r_kernel<<<8192, blk, 0, stream>>>(rhk, Rbb);
  // QR[z] = (q + r_r_bias) @ r_head_k^T, batched z = n*2+b (B shared per head)
  gemm_bf16_kernel<<<dim3(16, 8, 32), blk, 0, stream>>>(
      Qrb, Rbb, QRb, nullptr, nullptr, 64, 64, 64, 2048,
      (long)1024 * 64, (long)2048 * 64, 1, (long)1024 * 2048, 0);
  // fused attention: AC WMMA + shifted BD + mask + online softmax + P@V WMMA
  attn_kernel<<<dim3(8, 32, 1), blk, 0, stream>>>(Qwb, Khb, Vhb, QRb, avb);
  // attn_out = attn_vec @ o_w^T
  gemm_bf16_kernel<<<dim3(8, 16, 1), blk, 0, stream>>>(
      avb, owb, atno, nullptr, nullptr, 1024, 1024, 1024, 1024,
      0L, 0L, 0, 0L, 0);
  // h = LN(w + attn_out)
  add_ln_kernel<<<2048, blk, 0, stream>>>(w, atno, ln1_g, ln1_b, hbuf, hbf);
  // core1 = relu(h @ ff_w1^T + b1)  (bf16 mirror for next GEMM)
  gemm_bf16_kernel<<<dim3(32, 16, 1), blk, 0, stream>>>(
      hbf, f1wb, ff1o, coreb, ff_b1, 1024, 1024, 1024, 4096,
      0L, 0L, 0, 0L, 1);
  // core = core1 @ ff_w2^T + b2
  gemm_bf16_kernel<<<dim3(8, 16, 1), blk, 0, stream>>>(
      coreb, f2wb, ff2o, nullptr, ff_b2, 4096, 4096, 4096, 1024,
      0L, 0L, 0, 0L, 0);
  // out = LN(h + core)
  add_ln_kernel<<<2048, blk, 0, stream>>>(hbuf, ff2o, ln2_g, ln2_b, out, nullptr);
}
